// MambaBlock_42674795053653
// MI455X (gfx1250) — matrix operand order repaired and checked by reading, not compiled, not run on hardware
//
#include <hip/hip_runtime.h>
#include <hip/hip_bf16.h>

#define Bq 4
#define Tq 4096
#define Dq 1024
#define Nq 64
#define Rq 16

typedef __attribute__((ext_vector_type(16))) __bf16        v16bf;
typedef __attribute__((ext_vector_type(8)))  float         v8f;
typedef __attribute__((ext_vector_type(4)))  float         f32x4;
typedef __attribute__((ext_vector_type(4)))  unsigned int  u32x4;
typedef __attribute__((ext_vector_type(4)))  int           i32x4;

union BF16Frag { v16bf v; u32x4 q[2]; };

static __device__ __forceinline__ unsigned short f2bf_bits(float f) {
  __bf16 h = (__bf16)f;
  return __builtin_bit_cast(unsigned short, h);
}

static __device__ __forceinline__ void load8f(const float* p, float* o) {
  f32x4 a = *(const f32x4*)p;
  f32x4 c = *(const f32x4*)(p + 4);
  o[0] = a[0]; o[1] = a[1]; o[2] = a[2]; o[3] = a[3];
  o[4] = c[0]; o[5] = c[1]; o[6] = c[2]; o[7] = c[3];
}

// Load a prepacked B-fragment (32 lanes x 16 halfs, contiguous 32B per lane)
static __device__ __forceinline__ v16bf loadFrag(const unsigned short* p) {
  BF16Frag f;
  const u32x4* q = (const u32x4*)p;
  f.q[0] = q[0];
  f.q[1] = q[1];
  return f.v;
}

// ---- CDNA5 async global->LDS copy (16B granule), with portable fallback ----
static __device__ __forceinline__ void async_copy16(const void* gsrc, void* ldst) {
#if defined(__HIP_DEVICE_COMPILE__) && __has_builtin(__builtin_amdgcn_global_load_async_to_lds_b128)
  __builtin_amdgcn_global_load_async_to_lds_b128((i32x4*)gsrc, (i32x4*)ldst, 0, 0);
#else
  *(u32x4*)ldst = *(const u32x4*)gsrc;
#endif
}

static __device__ __forceinline__ void wait_async0() {
#if defined(__HIP_DEVICE_COMPILE__)
# if __has_builtin(__builtin_amdgcn_s_wait_asynccnt)
  __builtin_amdgcn_s_wait_asynccnt(0);
# else
  asm volatile("s_wait_asynccnt 0x0" ::: "memory");
# endif
#endif
}

// ---------------------------------------------------------------------------
// Kernel 0: weight repack into WMMA B-fragment layout (bf16).
// B layout for v_wmma_*_16x16x32: lane&15 = N (0..15), element i + lane half = K (0..31).
// ---------------------------------------------------------------------------
__global__ void __launch_bounds__(256) prep_kernel(
    const float* __restrict__ conv_w, const float* __restrict__ in_w,
    const float* __restrict__ dt_w,   const float* __restrict__ tr_w,
    const float* __restrict__ out_w,
    float* __restrict__ convT, unsigned short* __restrict__ in_pack,
    unsigned short* __restrict__ dt_pack, unsigned short* __restrict__ tr_pack,
    unsigned short* __restrict__ out_pack, float* __restrict__ zerobuf)
{
  int gid = blockIdx.x * blockDim.x + threadIdx.x;
  int nth = gridDim.x * blockDim.x;
  // convT[j][d]; fold the "+x" identity into tap j=3 (which multiplies x[t]).
  for (int d = gid; d < Dq; d += nth) {
    convT[0 * Dq + d] = conv_w[d * 4 + 0];
    convT[1 * Dq + d] = conv_w[d * 4 + 1];
    convT[2 * Dq + d] = conv_w[d * 4 + 2];
    convT[3 * Dq + d] = conv_w[d * 4 + 3] + 1.0f;
    zerobuf[d] = 0.0f;
  }
  // in_w_pack[c(32)][nt(4)][lane(32)][i(16)] = in_w[nt*16+(lane&15)][c*32+k], k = 16*(i>>3)+8*(lane>>4)+(i&7)
  for (int e = gid; e < 32 * 4 * 32 * 16; e += nth) {
    int i = e & 15, l = (e >> 4) & 31, nt = (e >> 9) & 3, c = e >> 11;
    in_pack[e] = f2bf_bits(in_w[(nt * 16 + (l & 15)) * Dq + c * 32 + ((i >> 3) << 4) + ((l >> 4) << 3) + (i & 7)]);
  }
  // dt_w_pack[c(32)][lane(32)][i(16)] = dt_w[lane&15][c*32+k], k = 16*(i>>3)+8*(lane>>4)+(i&7)
  for (int e = gid; e < 32 * 32 * 16; e += nth) {
    int i = e & 15, l = (e >> 4) & 31, c = e >> 9;
    dt_pack[e] = f2bf_bits(dt_w[(l & 15) * Dq + c * 32 + ((i >> 3) << 4) + ((l >> 4) << 3) + (i & 7)]);
  }
  // tr_w_pack[nt(4)][lane(32)][i(16)] = tr_w[nt*16+(lane&15)][k], k = 16*(i>>3)+8*(lane>>4)+(i&7) (K padded 16->32)
  for (int e = gid; e < 4 * 32 * 16; e += nth) {
    int i = e & 15, l = (e >> 4) & 31, nt = e >> 9;
    int k = ((i >> 3) << 4) + ((l >> 4) << 3) + (i & 7);
    tr_pack[e] = (k < Rq) ? f2bf_bits(tr_w[(nt * 16 + (l & 15)) * Rq + k])
                          : (unsigned short)0;
  }
  // out_w_pack[c(2)][nt(64)][lane(32)][i(16)] = out_w[nt*16+(lane&15)][c*32+k], k = 16*(i>>3)+8*(lane>>4)+(i&7)
  for (int e = gid; e < 2 * 64 * 32 * 16; e += nth) {
    int i = e & 15, l = (e >> 4) & 31, nt = (e >> 9) & 63, c = e >> 15;
    out_pack[e] = f2bf_bits(out_w[(nt * 16 + (l & 15)) * Nq + c * 32 + ((i >> 3) << 4) + ((l >> 4) << 3) + (i & 7)]);
  }
}

// ---------------------------------------------------------------------------
// Kernel 1: fused causal conv + projected GEMM + delta GEMM chain.
// One wave per 16-row (time) tile. K-loop over D in chunks of 32 (bf16 WMMA).
// A layout 16x32 bf16: lane = M (0..15, dup hi half), kb = 8*(lane>>4),
// element i<8 -> K = kb+i, i>=8 -> K = 16+kb+(i-8).
// Conv weights staged once per block into LDS (async); out-of-range conv rows
// read from a zero page -> branchless inner loop.
// ---------------------------------------------------------------------------
__global__ void __launch_bounds__(256) g1_kernel(
    const float* __restrict__ x, const float* __restrict__ convT,
    const unsigned short* __restrict__ in_pack,
    const unsigned short* __restrict__ dt_pack,
    const unsigned short* __restrict__ tr_pack,
    const float* __restrict__ in_b, const float* __restrict__ dt_b,
    const float* __restrict__ zerobuf,
    float* __restrict__ proj, float* __restrict__ delta)
{
  __shared__ __align__(16) float lds_cw[4 * Dq];           // 16 KB conv weights
  __shared__ __align__(32) unsigned short lds_h[8 * 256];  // tanh transpose
  const int wave = threadIdx.x >> 5;
  const int lane = threadIdx.x & 31;
  const int tile = blockIdx.x * 8 + wave;
  const int row  = lane & 15;
  const int half = lane >> 4;
  const int kb   = half * 8;
  const int gr0  = tile * 16;
  const int b    = gr0 / Tq;
  const int t0   = gr0 % Tq;
  const int t    = t0 + row;              // per-lane time index within batch b
  const float* xb = x + (size_t)b * Tq * Dq;

  // Stage convT (4*1024 f32 = 16KB) into LDS, 16B per copy, 4 per thread.
  for (int e = threadIdx.x; e < 4 * Dq / 4; e += 256)
    async_copy16(convT + e * 4, lds_cw + e * 4);
  wait_async0();
  __syncthreads();

  // Branchless conv row pointers: negative rows read the zero page.
  const float* xrow_j[3];
#pragma unroll
  for (int j = 0; j < 3; ++j) {
    const int tj = t - 3 + j;
    xrow_j[j] = (tj >= 0) ? (xb + (size_t)tj * Dq) : zerobuf;
  }
  const float* xrow_t = xb + (size_t)t * Dq;

  v8f accp[4];
  v8f acch = {0.f, 0.f, 0.f, 0.f, 0.f, 0.f, 0.f, 0.f};
#pragma unroll
  for (int nt = 0; nt < 4; ++nt) accp[nt] = acch;

  for (int c = 0; c < Dq / 32; ++c) {
    const int s0 = c * 32 + kb;           // elements 0..7
    const int s1 = s0 + 16;               // elements 8..15
    float xt[16];
    load8f(xrow_t + s0, xt);
    load8f(xrow_t + s1, xt + 8);
    float u[16];
    {
      float cw[16];
      load8f(lds_cw + 3 * Dq + s0, cw);
      load8f(lds_cw + 3 * Dq + s1, cw + 8);
#pragma unroll
      for (int i = 0; i < 16; ++i) u[i] = xt[i] * cw[i];
    }
#pragma unroll
    for (int j = 0; j < 3; ++j) {
      float cw[16], xs[16];
      load8f(lds_cw + j * Dq + s0, cw);
      load8f(lds_cw + j * Dq + s1, cw + 8);
      load8f(xrow_j[j] + s0, xs);
      load8f(xrow_j[j] + s1, xs + 8);
#pragma unroll
      for (int i = 0; i < 16; ++i) u[i] += xs[i] * cw[i];
    }
    BF16Frag au, ax;
#pragma unroll
    for (int i = 0; i < 16; ++i) {
      au.v[i] = (__bf16)u[i];
      ax.v[i] = (__bf16)xt[i];
    }
#pragma unroll
    for (int nt = 0; nt < 4; ++nt) {
      v16bf bf = loadFrag(in_pack + (((size_t)c * 4 + nt) * 32 + lane) * 16);
      accp[nt] = __builtin_amdgcn_wmma_f32_16x16x32_bf16(
          false, au.v, false, bf, (short)0, accp[nt], false, false);
    }
    {
      v16bf bf = loadFrag(dt_pack + ((size_t)c * 32 + lane) * 16);
      acch = __builtin_amdgcn_wmma_f32_16x16x32_bf16(
          false, ax.v, false, bf, (short)0, acch, false, false);
    }
  }

  // h = tanh(acch + dt_b); transpose C-layout -> A-layout through LDS (bf16).
  {
    const float db = dt_b[lane & 15];
    unsigned short* lh = lds_h + wave * 256;
#pragma unroll
    for (int v = 0; v < 8; ++v) {
      const int rm = v + half * 8;
      lh[rm * 16 + (lane & 15)] = f2bf_bits(tanhf(acch[v] + db));
    }
  }
  __syncthreads();

  BF16Frag ah;
  ah.q[0] = *(const u32x4*)(lds_h + wave * 256 + row * 16 + kb); // K = 0..15
  ah.q[1] = (u32x4){0u, 0u, 0u, 0u};                             // K = 16..31 pad

  v8f accd[4];
#pragma unroll
  for (int nt = 0; nt < 4; ++nt) {
    v8f z = {0.f, 0.f, 0.f, 0.f, 0.f, 0.f, 0.f, 0.f};
    v16bf bf = loadFrag(tr_pack + ((size_t)nt * 32 + lane) * 16);
    accd[nt] = __builtin_amdgcn_wmma_f32_16x16x32_bf16(
        false, ah.v, false, bf, (short)0, z, false, false);
  }

  // C layout store: element v -> row = v + 8*half, col = lane&15 (+16*nt)
#pragma unroll
  for (int nt = 0; nt < 4; ++nt) {
    const int ncol = nt * 16 + (lane & 15);
    const float bi = in_b[ncol];
#pragma unroll
    for (int v = 0; v < 8; ++v) {
      const int rm = v + half * 8;
      const size_t off = ((size_t)b * Tq + t0 + rm) * Nq + ncol;
      proj[off]  = accp[nt][v] + bi;
      delta[off] = accd[nt][v];
    }
  }
}

// ---------------------------------------------------------------------------
// Kernel 2: sequential nonlinear scan. 256 independent (b,n) chains.
// state = silu(delta + state) + projected; emit bf16 states for WMMA A use.
// ---------------------------------------------------------------------------
__global__ void __launch_bounds__(256) scan_kernel(
    const float* __restrict__ delta, const float* __restrict__ proj,
    unsigned short* __restrict__ states)
{
  const int i = threadIdx.x;
  const int b = i >> 6, n = i & 63;
  const size_t base = (size_t)b * Tq * Nq + n;
  const float* dp = delta + base;
  const float* pp = proj + base;
  unsigned short* sp = states + base;
  float s = 0.f;
#pragma unroll 4
  for (int t = 0; t < Tq; ++t) {
    const size_t o = (size_t)t * Nq;
    const float z = dp[o] + s;
    const float sig = __builtin_amdgcn_rcpf(1.f + __expf(-z));
    s = z * sig + pp[o];
    sp[o] = f2bf_bits(s);
  }
}

// ---------------------------------------------------------------------------
// Kernel 3: out = states @ out_w^T + out_b.  K = N = 64 -> 2 bf16 WMMA chunks.
// Block = one 16-ntile column group (256 cols) x 8 row tiles (one per wave).
// The group's 32KB of packed out_w is async-staged into LDS once per block.
// ---------------------------------------------------------------------------
__global__ void __launch_bounds__(256) g2_kernel(
    const unsigned short* __restrict__ states,
    const unsigned short* __restrict__ out_pack,
    const float* __restrict__ out_b, float* __restrict__ out)
{
  __shared__ __align__(32) unsigned short ldsB[2 * 16 * 512]; // 32 KB
  const int wave = threadIdx.x >> 5;
  const int lane = threadIdx.x & 31;
  const int cg   = blockIdx.x & 3;        // column group: 16 ntiles
  const int tg   = blockIdx.x >> 2;       // tile group: 8 row tiles
  const int tile = tg * 8 + wave;
  const int row  = lane & 15;
  const int half = lane >> 4;
  const int kb   = half * 8;
  const int gr0  = tile * 16;
  const int b    = gr0 / Tq;
  const int t0   = gr0 % Tq;

  // Stage this column group's B fragments: 2 chunks x 16 ntiles x 1KB.
  {
    const unsigned short* gc0 = out_pack + (size_t)(0 * 64 + cg * 16) * 512;
    const unsigned short* gc1 = out_pack + (size_t)(1 * 64 + cg * 16) * 512;
    for (int e = threadIdx.x; e < 2048; e += 256) {   // 2048 x 16B = 32KB
      const int c = e >> 10, off = (e & 1023) * 8;
      async_copy16((c ? gc1 : gc0) + off, ldsB + c * 8192 + off);
    }
  }
  wait_async0();
  __syncthreads();

  const unsigned short* srow = states + ((size_t)b * Tq + t0 + row) * Nq;
  BF16Frag a0, a1;
  a0.q[0] = *(const u32x4*)(srow + kb);            // K 0..15   (elems 0..7)
  a0.q[1] = *(const u32x4*)(srow + 16 + kb);       // K 16..31  (elems 8..15)
  a1.q[0] = *(const u32x4*)(srow + 32 + kb);       // K 32..47
  a1.q[1] = *(const u32x4*)(srow + 48 + kb);       // K 48..63

  float* orow = out + ((size_t)b * Tq + t0) * Dq;
#pragma unroll
  for (int ntl = 0; ntl < 16; ++ntl) {
    v16bf b0 = loadFrag(ldsB + (size_t)(0 * 16 + ntl) * 512 + lane * 16);
    v16bf b1 = loadFrag(ldsB + (size_t)(1 * 16 + ntl) * 512 + lane * 16);
    v8f acc = {0.f, 0.f, 0.f, 0.f, 0.f, 0.f, 0.f, 0.f};
    acc = __builtin_amdgcn_wmma_f32_16x16x32_bf16(false, a0.v, false, b0,
                                                  (short)0, acc, false, false);
    acc = __builtin_amdgcn_wmma_f32_16x16x32_bf16(false, a1.v, false, b1,
                                                  (short)0, acc, false, false);
    const int col = (cg * 16 + ntl) * 16 + (lane & 15);
    const float bi = out_b[col];
#pragma unroll
    for (int v = 0; v < 8; ++v) {
      const int rm = v + half * 8;
      orow[(size_t)rm * Dq + col] = acc[v] + bi;
    }
  }
}

// ---------------------------------------------------------------------------
extern "C" void kernel_launch(void* const* d_in, const int* in_sizes, int n_in,
                              void* d_out, int out_size, void* d_ws, size_t ws_size,
                              hipStream_t stream) {
  const float* x      = (const float*)d_in[0];
  const float* conv_w = (const float*)d_in[1];
  const float* in_w   = (const float*)d_in[2];
  const float* in_b   = (const float*)d_in[3];
  const float* dt_w   = (const float*)d_in[4];
  const float* dt_b   = (const float*)d_in[5];
  const float* tr_w   = (const float*)d_in[6];
  const float* out_w  = (const float*)d_in[7];
  const float* out_b  = (const float*)d_in[8];
  float* out = (float*)d_out;

  char* ws = (char*)d_ws;
  // Workspace layout (bytes):
  float*          convT    = (float*)(ws + 0);                  // 16 KB
  unsigned short* in_pack  = (unsigned short*)(ws + 16384);     // 128 KB
  unsigned short* dt_pack  = (unsigned short*)(ws + 147456);    // 32 KB
  unsigned short* tr_pack  = (unsigned short*)(ws + 180224);    // 4 KB
  unsigned short* out_pack = (unsigned short*)(ws + 184320);    // 128 KB
  float*          zerobuf  = (float*)(ws + 315392);             // 4 KB zero page
  float*          proj     = (float*)(ws + 327680);             // 4 MB
  float*          delta    = (float*)(ws + 327680 + 4194304);   // 4 MB
  unsigned short* states   = (unsigned short*)(ws + 327680 + 2 * 4194304); // 2 MB

  prep_kernel<<<dim3(64), dim3(256), 0, stream>>>(
      conv_w, in_w, dt_w, tr_w, out_w, convT, in_pack, dt_pack, tr_pack,
      out_pack, zerobuf);

  // 1024 row tiles, 8 waves per block -> 128 blocks
  g1_kernel<<<dim3(128), dim3(256), 0, stream>>>(
      x, convT, in_pack, dt_pack, tr_pack, in_b, dt_b, zerobuf, proj, delta);

  scan_kernel<<<dim3(1), dim3(256), 0, stream>>>(delta, proj, states);

  // 128 tile groups x 4 column groups -> 512 blocks
  g2_kernel<<<dim3(512), dim3(256), 0, stream>>>(states, out_pack, out_b, out);
}
